// SimpleGNN_1279900254387
// MI455X (gfx1250) — matrix operand order repaired and checked by reading, not compiled, not run on hardware
//
#include <hip/hip_runtime.h>
#include <hip/hip_bf16.h>
#include <math.h>

typedef __attribute__((ext_vector_type(16))) __bf16 v16bf;
typedef __attribute__((ext_vector_type(8)))  float  v8f;

struct bfchunks { uint4 lo; uint4 hi; };   // 32 bytes == v16bf

// ---------- bf16 helpers (native convert path, RNE) ----------
__device__ __forceinline__ unsigned short f2bfbits(float f) {
    __hip_bfloat16 b = __float2bfloat16(f);
    return __builtin_bit_cast(unsigned short, b);
}

__device__ __forceinline__ v8f wmma_bf16(v16bf a, v16bf b, v8f c) {
    // (neg_a, A, neg_b, B, c_mod, C, reuse_a, reuse_b)
    return __builtin_amdgcn_wmma_f32_16x16x32_bf16(false, a, false, b, (short)0, c,
                                                   false, false);
}

// ---------- kernel 1: zero workspace ----------
__global__ void gnn_zero_kernel(float* __restrict__ p, int n) {
    int i = blockIdx.x * blockDim.x + threadIdx.x;
    if (i < n) p[i] = 0.0f;
}

// ---------- kernel 2: edge scatter (deg + agg) ----------
__global__ void gnn_edge_kernel(const long long* __restrict__ ei,
                                const float* __restrict__ x,
                                float* __restrict__ deg,
                                float* __restrict__ agg,
                                int E) {
    int e = blockIdx.x * blockDim.x + threadIdx.x;
    if (e >= E) return;
    __builtin_prefetch(ei + e + 4096, 0, 0);            // global_prefetch_b8
    int s = (int)ei[e];
    int d = (int)ei[(size_t)E + e];
    atomicAdd(&deg[s], 1.0f);
    const float4* xr = (const float4*)(x + (size_t)d * 16);
    float4 v0 = xr[0], v1 = xr[1], v2 = xr[2], v3 = xr[3];
    float* ar = agg + (size_t)s * 16;
    atomicAdd(ar + 0,  v0.x); atomicAdd(ar + 1,  v0.y);
    atomicAdd(ar + 2,  v0.z); atomicAdd(ar + 3,  v0.w);
    atomicAdd(ar + 4,  v1.x); atomicAdd(ar + 5,  v1.y);
    atomicAdd(ar + 6,  v1.z); atomicAdd(ar + 7,  v1.w);
    atomicAdd(ar + 8,  v2.x); atomicAdd(ar + 9,  v2.y);
    atomicAdd(ar + 10, v2.z); atomicAdd(ar + 11, v2.w);
    atomicAdd(ar + 12, v3.x); atomicAdd(ar + 13, v3.y);
    atomicAdd(ar + 14, v3.z); atomicAdd(ar + 15, v3.w);
}

// ---------- kernel 3: fused 2-layer MLP via WMMA + column-sum ----------
// one wave = one 16-node tile; 8 waves / block
__global__ __launch_bounds__(256) void gnn_mlp_kernel(
        const float* __restrict__ x,   const float* __restrict__ agg,
        const float* __restrict__ deg,
        const float* __restrict__ W1,  const float* __restrict__ b1,
        const float* __restrict__ W2,  const float* __restrict__ b2,
        float* __restrict__ hsum, int n_nodes) {
    // weights staged TRANSPOSED: [col][k] so B-fragment K-runs are contiguous
    __shared__ __align__(16) unsigned short sW1T[64][32];     // K padded 16 -> 32
    __shared__ __align__(16) unsigned short sW2T[64][64];
    __shared__ float sb1[64], sb2[64], sHsum[64];
    __shared__ __align__(16) unsigned short sH1[8][16][64];   // per-wave h1 staging

    const int tid = threadIdx.x;

    for (int i = tid; i < 64 * 32; i += 256) {                // coalesced W1 read
        int k = i >> 6, c = i & 63;
        sW1T[c][k] = (k < 16) ? f2bfbits(W1[i]) : (unsigned short)0;
    }
    for (int i = tid; i < 64 * 64; i += 256) {                // coalesced W2 read
        int k = i >> 6, c = i & 63;
        sW2T[c][k] = f2bfbits(W2[i]);
    }
    if (tid < 64) { sb1[tid] = b1[tid]; sb2[tid] = b2[tid]; sHsum[tid] = 0.0f; }
    __syncthreads();

    const int wave = tid >> 5;
    const int lane = tid & 31;
    const int half = lane >> 4;      // 0: K-low half, 1: K-high half
    const int l16  = lane & 15;

    // --- preload B fragments with packed b128 loads (ISA B layout:
    //     lanes 0-15 hold K=k0+0..7,k0+16..23 at N=lane; lanes 16-31 K=k0+8..15,k0+24..31 at N=lane-16)
    v16bf B1[4];          // layer 1, K rows 16..31 are zero padding
    v16bf B2[2][4];       // layer 2, k0 = 0 / 32
    float bias1[4], bias2[4];
    for (int t = 0; t < 4; ++t) {
        const int col = t * 16 + l16;
        bias1[t] = sb1[col];
        bias2[t] = sb2[col];
        {
            const uint4* p = (const uint4*)&sW1T[col][half * 8];
            bfchunks ch; ch.lo = p[0]; ch.hi = p[2];          // K = half*8+0..7, 16+half*8+0..7
            B1[t] = __builtin_bit_cast(v16bf, ch);
        }
        for (int ks = 0; ks < 2; ++ks) {
            const uint4* p = (const uint4*)&sW2T[col][ks * 32 + half * 8];
            bfchunks ch; ch.lo = p[0]; ch.hi = p[2];
            B2[ks][t] = __builtin_bit_cast(v16bf, ch);
        }
    }

    float colsum[4] = {0.0f, 0.0f, 0.0f, 0.0f};   // per-lane columns are tile-invariant

    const int tiles = n_nodes >> 4;   // 100000/16 = 6250 exact
    for (int tile = blockIdx.x * 8 + wave; tile < tiles; tile += gridDim.x * 8) {
        const int node = (tile << 4) + l16;

        // ---- layer-1 A fragment: a = x + agg/deg (packed b128 loads) ----
        const float d   = fmaxf(deg[node], 1.0f);             // single v_max_num
        const float inv = __builtin_amdgcn_rcpf(d);           // v_rcp_f32: d is a count
        const float4* xp = (const float4*)(x   + (size_t)node * 16 + half * 8);
        const float4* ap = (const float4*)(agg + (size_t)node * 16 + half * 8);
        float4 x0 = xp[0], x1 = xp[1];
        float4 a0 = ap[0], a1 = ap[1];
        float av[8] = { x0.x + a0.x * inv, x0.y + a0.y * inv,
                        x0.z + a0.z * inv, x0.w + a0.w * inv,
                        x1.x + a1.x * inv, x1.y + a1.y * inv,
                        x1.z + a1.z * inv, x1.w + a1.w * inv };
        v16bf A;
        for (int i = 0; i < 8; ++i) {
            A[i]     = __builtin_bit_cast(__bf16, f2bfbits(av[i]));
            A[8 + i] = __builtin_bit_cast(__bf16, (unsigned short)0);  // K pad
        }

        // ---- layer 1: h1 = relu(A @ W1 + b1); bias pre-loaded into accumulator ----
        for (int t = 0; t < 4; ++t) {
            v8f c;
            for (int r = 0; r < 8; ++r) c[r] = bias1[t];      // broadcast bias as C
            c = wmma_bf16(A, B1[t], c);
            const int col = t * 16 + l16;
            for (int r = 0; r < 8; ++r) {
                float v = fmaxf(c[r], 0.0f);                  // single v_max_num ReLU
                sH1[wave][half * 8 + r][col] = f2bfbits(v);   // C layout: M = r + 8*half
            }
        }
        asm volatile("s_wait_dscnt 0" ::: "memory");  // intra-wave LDS RAW

        // ---- layer-2 A fragments: two aligned b128 chunks per K-step ----
        v16bf A2[2];
        for (int ks = 0; ks < 2; ++ks) {
            const uint4* p = (const uint4*)&sH1[wave][l16][ks * 32 + half * 8];
            bfchunks ch;
            ch.lo = p[0];          // K = k0 + half*8 + 0..7
            ch.hi = p[2];          // K = k0 + 16 + half*8 + 0..7  (32B further)
            A2[ks] = __builtin_bit_cast(v16bf, ch);
        }

        // ---- layer 2: h2 = relu(h1 @ W2 + b2); accumulate column sums ----
        for (int t = 0; t < 4; ++t) {
            v8f c;
            for (int r = 0; r < 8; ++r) c[r] = bias2[t];      // broadcast bias as C
            c = wmma_bf16(A2[0], B2[0][t], c);
            c = wmma_bf16(A2[1], B2[1][t], c);
            float s = 0.0f;
            for (int r = 0; r < 8; ++r)
                s += fmaxf(c[r], 0.0f);                       // single v_max_num ReLU
            colsum[t] += s;        // lanes l16 / l16+16 together cover all 16 rows
        }
    }

    for (int t = 0; t < 4; ++t)
        atomicAdd(&sHsum[t * 16 + l16], colsum[t]);   // once per thread, not per tile
    __syncthreads();
    if (tid < 64) atomicAdd(&hsum[tid], sHsum[tid]);
}

// ---------- kernel 4: mean + head + tanh ----------
__global__ void gnn_head_kernel(const float* __restrict__ hsum,
                                const float* __restrict__ Wh,
                                const float* __restrict__ bh,
                                float* __restrict__ out, float inv_n) {
    __shared__ float acc[64];
    const int t = threadIdx.x;
    acc[t] = hsum[t] * inv_n * Wh[t];
    __syncthreads();
    for (int s = 32; s > 0; s >>= 1) {
        if (t < s) acc[t] += acc[t + s];
        __syncthreads();
    }
    if (t == 0) out[0] = tanhf(acc[0] + bh[0]);
}

extern "C" void kernel_launch(void* const* d_in, const int* in_sizes, int n_in,
                              void* d_out, int out_size, void* d_ws, size_t ws_size,
                              hipStream_t stream) {
    const float*     x  = (const float*)d_in[0];
    const long long* ei = (const long long*)d_in[1];
    const float*     W1 = (const float*)d_in[2];
    const float*     b1 = (const float*)d_in[3];
    const float*     W2 = (const float*)d_in[4];
    const float*     b2 = (const float*)d_in[5];
    const float*     Wh = (const float*)d_in[6];
    const float*     bh = (const float*)d_in[7];
    float*           out = (float*)d_out;

    const int n = in_sizes[0] / 16;     // 100000 nodes
    const int E = in_sizes[1] / 2;      // 3200000 edges

    char*  ws   = (char*)d_ws;
    float* deg  = (float*)ws;                                             // n
    float* agg  = (float*)(ws + (size_t)n * sizeof(float));               // n*16
    float* hsum = (float*)(ws + (size_t)n * 17 * sizeof(float));          // 64

    const int zn = n * 17 + 64;
    gnn_zero_kernel<<<(zn + 255) / 256, 256, 0, stream>>>(deg, zn);
    gnn_edge_kernel<<<(E + 255) / 256, 256, 0, stream>>>(ei, x, deg, agg, E);

    const int tiles  = n / 16;
    const int blocks = (tiles + 7) / 8;
    gnn_mlp_kernel<<<blocks, 256, 0, stream>>>(x, agg, deg, W1, b1, W2, b2, hsum, n);
    gnn_head_kernel<<<1, 64, 0, stream>>>(hsum, Wh, bh, out, 1.0f / (float)n);
}
